// ResRnn_82222853914976
// MI455X (gfx1250) — matrix operand order repaired and checked by reading, not compiled, not run on hardware
//
#include <hip/hip_runtime.h>

// ---------------- problem constants ----------------
#define SEQ   2048
#define BATCH 32
#define IW    64      // INPUT_WIDTH
#define STW   960     // STATE_WIDTH
#define SW    1024    // STREAM_WIDTH
#define OW    64      // OUTPUT_WIDTH

#define NBLK  32      // persistent workgroups (grid)
#define CPB   32      // stream columns per workgroup
#define TPB   128     // 4 waves of 32
#define LDSST 1032    // padded LDS row stride for weights (elements)

// TDM-staged A matrix: rows of 2048B data + 2x16B pad -> 2080B stride
#define ASTRIDE_B 2080

typedef __attribute__((ext_vector_type(16))) __bf16 v16bf;
typedef __attribute__((ext_vector_type(8)))  __bf16 v8bf;
typedef __attribute__((ext_vector_type(8)))  float  v8f;
typedef unsigned int u32x4 __attribute__((ext_vector_type(4)));
typedef int          i32x8 __attribute__((ext_vector_type(8)));
typedef int          i32x4 __attribute__((ext_vector_type(4)));

struct Bar { unsigned cnt; unsigned gen; };

// ---- workspace layout (in d_ws) ----
#define WS_SF32_OFF  256
#define WS_SBF_OFF   (WS_SF32_OFF + 2*BATCH*SW*4)
#define WS_HBUF_OFF  (WS_SBF_OFF + 2*BATCH*SW*2)

__device__ __forceinline__ void gridbar(Bar* bar) {
  __syncthreads();
  if (threadIdx.x == 0) {
    unsigned gen  = __hip_atomic_load(&bar->gen, __ATOMIC_RELAXED, __HIP_MEMORY_SCOPE_AGENT);
    unsigned prev = __hip_atomic_fetch_add(&bar->cnt, 1u, __ATOMIC_ACQ_REL, __HIP_MEMORY_SCOPE_AGENT);
    if (prev == NBLK - 1u) {
      __hip_atomic_store(&bar->cnt, 0u, __ATOMIC_RELAXED, __HIP_MEMORY_SCOPE_AGENT);
      __hip_atomic_fetch_add(&bar->gen, 1u, __ATOMIC_RELEASE, __HIP_MEMORY_SCOPE_AGENT);
    } else {
      while (__hip_atomic_load(&bar->gen, __ATOMIC_ACQUIRE, __HIP_MEMORY_SCOPE_AGENT) == gen) {
        __builtin_amdgcn_s_sleep(2);
      }
    }
  }
  __syncthreads();
}

// Build v16bf from two 16B chunks (A: chunks at +0 and +32B; B: +0 and +16 elems (32B))
__device__ __forceinline__ v16bf frag2(const v8bf* p0, const v8bf* p1) {
  v8bf lo = *p0, hi = *p1;
  v16bf r;
#pragma unroll
  for (int i = 0; i < 8; ++i) { r[i] = lo[i]; r[i + 8] = hi[i]; }
  return r;
}

// TDM: DMA a dense [32 x 1024] bf16 tile (global, row stride 1024 elems) into LDS
// with 16B padding every 1024B -> LDS row stride 2080B. Issue once per WG (wave 0).
__device__ __forceinline__ void tdm_stage_A(unsigned lds_addr, const __bf16* src) {
  unsigned long long ga = (unsigned long long)(uintptr_t)src;
  u32x4 g0;
  g0[0] = 1u;                                    // count=1, no gather
  g0[1] = lds_addr;                              // lds_addr[31:0]
  g0[2] = (unsigned)(ga & 0xFFFFFFFFu);          // global_addr[31:0]
  g0[3] = (unsigned)((ga >> 32) & 0x01FFFFFFu)   // global_addr[56:32]
        | (2u << 30);                            // type=2 (image)
  i32x8 g1;
  g1[0] = (1 << 16)                              // data_size = 2 bytes
        | (1 << 20)                              // pad_enable
        | (7 << 22)                              // pad_interval: 256 DWORDs (1024B)
        | (3 << 25);                             // pad_amount: 4 DWORDs (16B)
  g1[1] = (SW & 0xFFFF) << 16;                   // tensor_dim0[15:0]=1024 (lo bits in [31:16])
  g1[2] = (SW >> 16) | (BATCH << 16);            // tensor_dim0[31:16]=0 | tensor_dim1[15:0]=32
  g1[3] = (SW << 16);                            // tensor_dim1 hi=0 | tile_dim0=1024
  g1[4] = BATCH;                                 // tile_dim1=32 | tile_dim2=0
  g1[5] = SW;                                    // tensor_dim0_stride[31:0]=1024
  g1[6] = 0;                                     // dim0_stride hi | dim1_stride lo
  g1[7] = 0;                                     // dim1_stride hi
  i32x4 z4 = {0, 0, 0, 0};                       // groups 2/3 unused (2-D tensor)
  i32x8 z8 = {0, 0, 0, 0, 0, 0, 0, 0};
  __builtin_amdgcn_tensor_load_to_lds(g0, g1, z4, z4, z8, 0);
}

// ---------------- init: barrier + s0 = [x_0 | (1-lin)*initial_state] ----------------
__global__ void rnn_init_kernel(const float* __restrict__ inp,
                                const float* __restrict__ istate,
                                char* __restrict__ ws) {
  Bar* bar = (Bar*)ws;
  if (blockIdx.x == 0 && threadIdx.x == 0) { bar->cnt = 0u; bar->gen = 0u; }
  float*  sf32 = (float*)(ws + WS_SF32_OFF);
  __bf16* sbf  = (__bf16*)(ws + WS_SBF_OFF);
  const float onem = (float)(1.0 - 0.99999);
  int idx = blockIdx.x * blockDim.x + threadIdx.x;
  for (int i = idx; i < BATCH * SW; i += gridDim.x * blockDim.x) {
    int b = i >> 10, c = i & (SW - 1);
    float v = (c < IW) ? inp[b * IW + c] : onem * istate[c - IW];
    sf32[i] = v;
    sbf[i]  = (__bf16)v;
  }
}

// ---------------- persistent RNN kernel ----------------
__global__ __launch_bounds__(TPB, 1)
void rnn_persist_kernel(const float* __restrict__ inp,
                        const float* __restrict__ W1, const float* __restrict__ b1,
                        const float* __restrict__ W2, const float* __restrict__ b2,
                        float* __restrict__ out, char* __restrict__ ws) {
  __shared__ __bf16 w1s[CPB * LDSST];                                     // 64.5KB
  __shared__ __bf16 w2s[CPB * LDSST];                                     // 64.5KB
  __shared__ __attribute__((aligned(16))) char astage[BATCH * ASTRIDE_B]; // 65KB

  Bar*    bar  = (Bar*)ws;
  float*  sf32 = (float*)(ws + WS_SF32_OFF);
  __bf16* sbf  = (__bf16*)(ws + WS_SBF_OFF);
  __bf16* hbuf = (__bf16*)(ws + WS_HBUF_OFF);

  const int tid   = threadIdx.x;
  const int lane  = tid & 31;
  const int wv    = tid >> 5;
  const int ln    = lane & 15;
  const int lh    = lane >> 4;
  const int j0    = blockIdx.x * CPB;
  const int mtile = (wv >> 1) << 4;   // 0 or 16
  const int ntile = (wv & 1) << 4;    // 0 or 16

  // ---- preload weight slices into LDS as bf16 ----
  for (int i = tid; i < CPB * SW; i += TPB) {
    int jr = i >> 10, k = i & (SW - 1);
    w1s[jr * LDSST + k] = (__bf16)W1[(size_t)(j0 + jr) * SW + k];
    w2s[jr * LDSST + k] = (__bf16)W2[(size_t)(j0 + jr) * SW + k];
  }

  const unsigned astage_lds = (unsigned)(uintptr_t)&astage[0];
  const int   jcol  = j0 + ntile + ln;
  const float bias1 = b1[jcol];
  const float bias2 = b2[jcol];
  const __bf16* brow1 = w1s + (ntile + ln) * LDSST;
  const __bf16* brow2 = w2s + (ntile + ln) * LDSST;
  // A row base in staged LDS for this lane's M rows
  const char* aRowS = &astage[(mtile + ln) * ASTRIDE_B];
  const int   akB   = lh * 16;   // A lane-half K offset in bytes (8 elems)
  const int   bk    = lh * 8;    // B lane-half K offset in elements

  const float kLin = 0.99999f;
  const float kOne = (float)(1.0 - 0.99999);

  float* outO = out;
  float* outS = out + (size_t)SEQ * BATCH * OW;

  __syncthreads();

  for (int t = 0; t < SEQ; ++t) {
    const int cur = t & 1, nxt = cur ^ 1;

    // ---- stage s[cur] into LDS via Tensor Data Mover ----
    if (wv == 0) {
      tdm_stage_A(astage_lds, sbf + cur * BATCH * SW);
      __builtin_amdgcn_s_wait_tensorcnt(0);
    }
    __syncthreads();

    // ---- phase 1: h = relu(s @ W1^T + b1) ----
    v8f c;
#pragma unroll
    for (int r = 0; r < 8; ++r) c[r] = bias1;
    // K in [0,512): no mid-row pad adjustment
#pragma unroll 4
    for (int k0 = 0; k0 < SW / 2; k0 += 32) {
      v16bf a  = frag2((const v8bf*)(aRowS + k0 * 2 + akB),
                       (const v8bf*)(aRowS + k0 * 2 + akB + 32));
      v16bf bm = frag2((const v8bf*)(brow1 + k0 + bk),
                       (const v8bf*)(brow1 + k0 + bk + 16));
      c = __builtin_amdgcn_wmma_f32_16x16x32_bf16(false, a, false, bm, (short)0, c, false, false);
    }
    // K in [512,1024): +16B LDS pad offset
#pragma unroll 4
    for (int k0 = SW / 2; k0 < SW; k0 += 32) {
      v16bf a  = frag2((const v8bf*)(aRowS + k0 * 2 + akB + 16),
                       (const v8bf*)(aRowS + k0 * 2 + akB + 48));
      v16bf bm = frag2((const v8bf*)(brow1 + k0 + bk),
                       (const v8bf*)(brow1 + k0 + bk + 16));
      c = __builtin_amdgcn_wmma_f32_16x16x32_bf16(false, a, false, bm, (short)0, c, false, false);
    }
    {
      const int m0 = mtile + (lh << 3);
#pragma unroll
      for (int r = 0; r < 8; ++r) {
        float h = c[r] > 0.0f ? c[r] : 0.0f;
        hbuf[(m0 + r) * SW + jcol] = (__bf16)h;
      }
    }
    gridbar(bar);

    // ---- stage h into LDS via TDM ----
    if (wv == 0) {
      tdm_stage_A(astage_lds, hbuf);
      __builtin_amdgcn_s_wait_tensorcnt(0);
    }
    __syncthreads();

    // ---- phase 2: o = h @ W2^T + b2 ; stream = lin*s + (1-lin)*o ----
#pragma unroll
    for (int r = 0; r < 8; ++r) c[r] = bias2;
#pragma unroll 4
    for (int k0 = 0; k0 < SW / 2; k0 += 32) {
      v16bf a  = frag2((const v8bf*)(aRowS + k0 * 2 + akB),
                       (const v8bf*)(aRowS + k0 * 2 + akB + 32));
      v16bf bm = frag2((const v8bf*)(brow2 + k0 + bk),
                       (const v8bf*)(brow2 + k0 + bk + 16));
      c = __builtin_amdgcn_wmma_f32_16x16x32_bf16(false, a, false, bm, (short)0, c, false, false);
    }
#pragma unroll 4
    for (int k0 = SW / 2; k0 < SW; k0 += 32) {
      v16bf a  = frag2((const v8bf*)(aRowS + k0 * 2 + akB + 16),
                       (const v8bf*)(aRowS + k0 * 2 + akB + 48));
      v16bf bm = frag2((const v8bf*)(brow2 + k0 + bk),
                       (const v8bf*)(brow2 + k0 + bk + 16));
      c = __builtin_amdgcn_wmma_f32_16x16x32_bf16(false, a, false, bm, (short)0, c, false, false);
    }
    {
      const float*  sc  = sf32 + cur * BATCH * SW;
      float*        sn  = sf32 + nxt * BATCH * SW;
      __bf16*       snb = sbf  + nxt * BATCH * SW;
      const int m0 = mtile + (lh << 3);
#pragma unroll
      for (int r = 0; r < 8; ++r) {
        const int m = m0 + r;
        float stream = kLin * sc[m * SW + jcol] + kOne * c[r];
        if (jcol >= STW) {
          outO[(size_t)t * BATCH * OW + (size_t)m * OW + (jcol - STW)] = stream;
        } else {
          outS[(size_t)t * BATCH * STW + (size_t)m * STW + jcol] = stream;
          sn [m * SW + (IW + jcol)] = stream;
          snb[m * SW + (IW + jcol)] = (__bf16)stream;
        }
      }
      if (blockIdx.x == 0 && (t + 1) < SEQ) {
        const float* xn = inp + (size_t)(t + 1) * BATCH * IW;
        for (int i = tid; i < BATCH * IW; i += TPB) {
          int b = i >> 6, cc = i & (IW - 1);
          float v = xn[i];
          sn [b * SW + cc] = v;
          snb[b * SW + cc] = (__bf16)v;
        }
      }
    }
    gridbar(bar);
  }
}

extern "C" void kernel_launch(void* const* d_in, const int* in_sizes, int n_in,
                              void* d_out, int out_size, void* d_ws, size_t ws_size,
                              hipStream_t stream) {
  const float* inp    = (const float*)d_in[0];
  const float* istate = (const float*)d_in[1];
  const float* W1     = (const float*)d_in[2];
  const float* b1     = (const float*)d_in[3];
  const float* W2     = (const float*)d_in[4];
  const float* b2     = (const float*)d_in[5];
  float* out = (float*)d_out;
  char*  ws  = (char*)d_ws;

  rnn_init_kernel<<<NBLK, TPB, 0, stream>>>(inp, istate, ws);
  rnn_persist_kernel<<<NBLK, TPB, 0, stream>>>(inp, W1, b1, W2, b2, out, ws);
}
